// MultiHeadAttention_37177236914630
// MI455X (gfx1250) — matrix operand order repaired and checked by reading, not compiled, not run on hardware
//
#include <hip/hip_runtime.h>

// ---------------- problem constants ----------------
#define Bsz 4
#define Tn  2048
#define Dm  1024
#define Hn  16
#define DHn 64
#define M_ROWS (Bsz * Tn)        // 8192
#define N_QKV  (3 * Dm)          // 3072

// ---------------- feature detection: CDNA5 async global->LDS ----------------
#if defined(__has_builtin)
#if __has_builtin(__builtin_amdgcn_global_load_async_to_lds_b128)
#define HAVE_ASYNC_LDS 1
#endif
#endif
#ifndef HAVE_ASYNC_LDS
#define HAVE_ASYNC_LDS 0
#endif

// ---------------- vector types for WMMA ----------------
typedef __attribute__((ext_vector_type(16))) __bf16   v16bf;
typedef __attribute__((ext_vector_type(8)))  float    v8f;
typedef __attribute__((ext_vector_type(4)))  unsigned uint4v;

#if HAVE_ASYNC_LDS
// builtin expects pointer-to-<4 x i32> in AS1 (global) / AS3 (LDS)
typedef int b128_t __attribute__((vector_size(16)));
typedef __attribute__((address_space(1))) b128_t* gas_b128p;
typedef __attribute__((address_space(3))) b128_t* las_b128p;
#endif

union B16x16 {              // 16 bf16 = 32 bytes, loaded as two 16B pieces
    v16bf  v;
    uint4v q[2];
    unsigned short s[16];
};

__device__ __forceinline__ unsigned short f2bf(float f) {
    unsigned u = __float_as_uint(f);
    unsigned r = u + 0x7FFFu + ((u >> 16) & 1u);   // round-to-nearest-even
    return (unsigned short)(r >> 16);
}

__device__ __forceinline__ v8f wmma_bf16(const B16x16& a, const B16x16& b, v8f c) {
    return __builtin_amdgcn_wmma_f32_16x16x32_bf16(
        false, a.v, false, b.v, (short)0, c, false, false);
}

// ---------------- conversion kernels ----------------
__global__ void cvt_f32_bf16(const float* __restrict__ src,
                             unsigned short* __restrict__ dst, int n) {
    int i = blockIdx.x * blockDim.x + threadIdx.x;
    if (i < n) dst[i] = f2bf(src[i]);
}

// Wt[c][r] = W[r][c]  (so GEMM B-operand reads are K-contiguous)
__global__ void transpose_bf16(const float* __restrict__ W,
                               unsigned short* __restrict__ Wt, int R, int C) {
    int i = blockIdx.x * blockDim.x + threadIdx.x;
    if (i < R * C) {
        int r = i / C, c = i % C;
        Wt[(size_t)c * R + r] = f2bf(W[i]);
    }
}

// ---------------- cooperative GEMM core ----------------
// Block = 256 threads = 8 waves. Block tile: 256 (M) x 64 (N).
// Each wave owns a 32x64 sub-tile (acc[2][4] C tiles).
// B-panel (64 cols x 32 K, bf16, 4KB) staged into LDS by the whole block,
// via CDNA5 async global->LDS when available.
__device__ __forceinline__ void gemm_core(const unsigned short* __restrict__ A,
                                          const unsigned short* __restrict__ Wt,
                                          unsigned short* Bs,    // LDS, 64*32 u16
                                          int m0w, int n0, int lane,
                                          v8f acc[2][4]) {
    const int r = lane & 15, hl = lane >> 4;
    const int tid   = threadIdx.x;
    const int j     = tid >> 2;        // 0..63 : B row (output column)
    const int piece = tid & 3;         // 0..3  : 8-element chunk of 32 K values

    const unsigned short* a0row = A + (size_t)(m0w + r) * Dm;
    const unsigned short* a1row = A + (size_t)(m0w + 16 + r) * Dm;
    const unsigned short* gbase = Wt + (size_t)(n0 + j) * Dm + piece * 8;
    unsigned short*       lbase = Bs + j * 32 + piece * 8;

    for (int k0 = 0; k0 < Dm; k0 += 32) {
        // ---- stage B panel: Wt[n0..n0+63][k0..k0+31] -> LDS
        const unsigned short* g = gbase + k0;
#if HAVE_ASYNC_LDS
        __builtin_amdgcn_global_load_async_to_lds_b128(
            (gas_b128p)g, (las_b128p)lbase, 0, 0);
        asm volatile("s_wait_asynccnt 0" ::: "memory");
#else
        *(uint4v*)lbase = *(const uint4v*)g;
#endif
        __syncthreads();

        // ---- A operands (two 16-row tiles), global
        B16x16 a0, a1;
        a0.q[0] = *(const uint4v*)(a0row + k0 + hl * 8);
        a0.q[1] = *(const uint4v*)(a0row + k0 + 16 + hl * 8);
        a1.q[0] = *(const uint4v*)(a1row + k0 + hl * 8);
        a1.q[1] = *(const uint4v*)(a1row + k0 + 16 + hl * 8);

        // ---- 8 WMMAs per k-step from the shared B panel
        for (int t = 0; t < 4; ++t) {
            const unsigned short* bl = Bs + (t * 16 + r) * 32 + hl * 8;
            B16x16 bm;
            bm.q[0] = *(const uint4v*)(bl);
            bm.q[1] = *(const uint4v*)(bl + 16);
            acc[0][t] = wmma_bf16(a0, bm, acc[0][t]);
            acc[1][t] = wmma_bf16(a1, bm, acc[1][t]);
        }
        __syncthreads();   // protect LDS before next stage overwrites
    }
}

// ---------------- QKV projection GEMM ----------------
// Scatter to Q/K (b,h,t,dh) and V transposed (b,h,dh,t).
__global__ void qkv_gemm_kernel(const unsigned short* __restrict__ Xb,
                                const unsigned short* __restrict__ Wt,
                                const float* __restrict__ bias,
                                unsigned short* __restrict__ Qo,
                                unsigned short* __restrict__ Ko,
                                unsigned short* __restrict__ Vto) {
    __shared__ unsigned short Bs[64 * 32];
    const int Ngrp = N_QKV / 64;       // 48
    int mb = blockIdx.x / Ngrp, ng = blockIdx.x % Ngrp;
    int lane = threadIdx.x & 31, r = lane & 15, hl = lane >> 4;
    int m0w = mb * 256 + (threadIdx.x >> 5) * 32;
    int n0  = ng * 64;

    v8f acc[2][4];
    for (int m = 0; m < 2; ++m)
        for (int t = 0; t < 4; ++t)
            for (int i = 0; i < 8; ++i) acc[m][t][i] = 0.0f;

    gemm_core(Xb, Wt, Bs, m0w, n0, lane, acc);

    for (int t = 0; t < 4; ++t) {
        int n = n0 + t * 16 + r;
        float bv = bias[n];
        int sel = n / Dm, c = n % Dm, h = c / DHn, dh = c % DHn;
        for (int mi = 0; mi < 2; ++mi) {
            for (int v = 0; v < 8; ++v) {
                int m = m0w + mi * 16 + v + 8 * hl;
                int b = m >> 11, tt = m & (Tn - 1);
                unsigned short val = f2bf(acc[mi][t][v] + bv);
                size_t bh = (size_t)(b * Hn + h);
                if (sel == 0)      Qo[(bh * Tn + tt) * DHn + dh] = val;
                else if (sel == 1) Ko[(bh * Tn + tt) * DHn + dh] = val;
                else               Vto[(bh * DHn + dh) * Tn + tt] = val;
            }
        }
    }
}

// ---------------- output projection GEMM ----------------
__global__ void out_gemm_kernel(const unsigned short* __restrict__ Ab,
                                const unsigned short* __restrict__ Wt,
                                const float* __restrict__ bias,
                                float* __restrict__ out) {
    __shared__ unsigned short Bs[64 * 32];
    const int Ngrp = Dm / 64;          // 16
    int mb = blockIdx.x / Ngrp, ng = blockIdx.x % Ngrp;
    int lane = threadIdx.x & 31, r = lane & 15, hl = lane >> 4;
    int m0w = mb * 256 + (threadIdx.x >> 5) * 32;
    int n0  = ng * 64;

    v8f acc[2][4];
    for (int m = 0; m < 2; ++m)
        for (int t = 0; t < 4; ++t)
            for (int i = 0; i < 8; ++i) acc[m][t][i] = 0.0f;

    gemm_core(Ab, Wt, Bs, m0w, n0, lane, acc);

    for (int t = 0; t < 4; ++t) {
        int n = n0 + t * 16 + r;
        float bv = bias[n];
        for (int mi = 0; mi < 2; ++mi)
            for (int v = 0; v < 8; ++v) {
                int m = m0w + mi * 16 + v + 8 * hl;
                out[(size_t)m * Dm + n] = acc[mi][t][v] + bv;
            }
    }
}

// ---------------- flash attention (1 wave per 16-row q-tile) ----------------
__global__ void attn_kernel(const unsigned short* __restrict__ Q,
                            const unsigned short* __restrict__ K,
                            const unsigned short* __restrict__ Vt,
                            const unsigned char* __restrict__ mask,
                            float* __restrict__ head_out,
                            unsigned short* __restrict__ attn_bf16) {
    __shared__ unsigned short pl[8][16 * 32];    // per-wave P relayout buffer
    int wave = (blockIdx.x * blockDim.x + threadIdx.x) >> 5;
    const int nQT = Tn / 16;                     // 128
    if (wave >= Bsz * Hn * nQT) return;
    int lane = threadIdx.x & 31, r = lane & 15, hl = lane >> 4;
    int qt = wave % nQT, bh = wave / nQT;
    int h = bh % Hn, b = bh / Hn;
    int q0 = qt * 16;
    unsigned short* myP = pl[threadIdx.x >> 5];

    // Q A-operands for dh chunks [0,32) and [32,64)
    const unsigned short* Qbase = Q + ((size_t)bh * Tn + q0) * DHn;
    B16x16 qa[2];
    for (int c = 0; c < 2; ++c) {
        const unsigned short* qr = Qbase + r * DHn + c * 32;
        qa[c].q[0] = *(const uint4v*)(qr + hl * 8);
        qa[c].q[1] = *(const uint4v*)(qr + 16 + hl * 8);
    }

    float mrow[8], lrow[8];
    v8f acc[4];
    for (int v = 0; v < 8; ++v) { mrow[v] = -1e30f; lrow[v] = 0.0f; }
    for (int t = 0; t < 4; ++t) for (int i = 0; i < 8; ++i) acc[t][i] = 0.0f;

    const unsigned short* Kbase = K  + (size_t)bh * Tn * DHn;
    const unsigned short* Vbase = Vt + (size_t)bh * DHn * Tn;
    const unsigned char*  mk    = mask + (size_t)b * Tn;
    const float scale = 0.125f;                  // 1/sqrt(64)
    int nKB = (q0 + 16 + 31) >> 5;               // causal: only k-blocks <= diag

    for (int kb = 0; kb < nKB; ++kb) {
        int k0 = kb * 32;
        // ---- S = Q . K^T : two 16x16 C tiles, K-dim = 64 (2 WMMAs each)
        v8f s[2];
        for (int st = 0; st < 2; ++st) {
            v8f c; for (int i = 0; i < 8; ++i) c[i] = 0.0f;
            int key = k0 + st * 16 + r;          // B-operand column = key row
            const unsigned short* kr = Kbase + (size_t)key * DHn;
            for (int dc = 0; dc < 2; ++dc) {
                B16x16 bm;
                bm.q[0] = *(const uint4v*)(kr + dc * 32 + hl * 8);
                bm.q[1] = *(const uint4v*)(kr + dc * 32 + 16 + hl * 8);
                c = wmma_bf16(qa[dc], bm, c);
            }
            s[st] = c;
        }
        // ---- scale + causal + pad mask (C layout: M = v + 8*hl, N = r)
        for (int st = 0; st < 2; ++st) {
            int kcol = k0 + st * 16 + r;
            bool kvalid = (mk[kcol] != 0);
            for (int v = 0; v < 8; ++v) {
                int qrow = q0 + v + 8 * hl;
                bool ok = kvalid && (kcol <= qrow);
                s[st][v] = ok ? s[st][v] * scale : -1e30f;
            }
        }
        // ---- online softmax: row stats per-(vgpr, 16-lane half)
        float anew[8];
        for (int v = 0; v < 8; ++v) {
            float x = fmaxf(s[0][v], s[1][v]);
            for (int msk = 1; msk < 16; msk <<= 1)
                x = fmaxf(x, __shfl_xor(x, msk, 32));
            float mnew = fmaxf(mrow[v], x);
            anew[v] = __expf(mrow[v] - mnew);
            mrow[v] = mnew;
        }
        for (int v = 0; v < 8; ++v) {
            float p0 = __expf(s[0][v] - mrow[v]);
            float p1 = __expf(s[1][v] - mrow[v]);
            float rs = p0 + p1;
            for (int msk = 1; msk < 16; msk <<= 1)
                rs += __shfl_xor(rs, msk, 32);
            lrow[v] = lrow[v] * anew[v] + rs;
            int prow = v + 8 * hl;
            myP[prow * 32 + r]      = f2bf(p0);   // C layout -> row-major LDS
            myP[prow * 32 + 16 + r] = f2bf(p1);
        }
        for (int t = 0; t < 4; ++t)
            for (int v = 0; v < 8; ++v) acc[t][v] *= anew[v];
        // same-wave LDS write->read: force DS completion, block reordering
        asm volatile("s_wait_dscnt 0" ::: "memory");
        // ---- reload P in A-operand layout
        B16x16 pa;
        pa.q[0] = *(const uint4v*)(myP + r * 32 + hl * 8);
        pa.q[1] = *(const uint4v*)(myP + r * 32 + 16 + hl * 8);
        // ---- acc += P . V   (Vt rows are key-contiguous)
        for (int t = 0; t < 4; ++t) {
            int dh = t * 16 + r;
            const unsigned short* vr = Vbase + (size_t)dh * Tn + k0 + hl * 8;
            B16x16 bm;
            bm.q[0] = *(const uint4v*)(vr);
            bm.q[1] = *(const uint4v*)(vr + 16);
            acc[t] = wmma_bf16(pa, bm, acc[t]);
        }
    }
    // ---- epilogue: normalize, write head_outputs (f32) + bf16 copy
    float* ho = head_out + ((size_t)bh * Tn + q0) * DHn;
    for (int t = 0; t < 4; ++t) {
        int dh = t * 16 + r;
        for (int v = 0; v < 8; ++v) {
            int qrow = v + 8 * hl;
            float o = acc[t][v] / lrow[v];
            ho[(size_t)qrow * DHn + dh] = o;
            int tg = q0 + qrow;
            attn_bf16[((size_t)(b * Tn + tg)) * Dm + h * DHn + dh] = f2bf(o);
        }
    }
}

// ---------------- host launch ----------------
extern "C" void kernel_launch(void* const* d_in, const int* in_sizes, int n_in,
                              void* d_out, int out_size, void* d_ws, size_t ws_size,
                              hipStream_t stream) {
    (void)in_sizes; (void)n_in; (void)out_size; (void)ws_size;
    const float*         x    = (const float*)d_in[0];
    const unsigned char* mask = (const unsigned char*)d_in[1];
    const float*         Wqkv = (const float*)d_in[2];
    const float*         bqkv = (const float*)d_in[3];
    const float*         Wout = (const float*)d_in[4];
    const float*         bout = (const float*)d_in[5];

    float* out      = (float*)d_out;
    float* head_out = out + (size_t)Bsz * Tn * Dm;   // second tuple element

    char*  ws  = (char*)d_ws;
    size_t off = 0;
    auto alloc = [&](size_t bytes) -> void* {
        void* p = ws + off;
        off += (bytes + 255) & ~(size_t)255;
        return p;
    };
    unsigned short* Xb    = (unsigned short*)alloc((size_t)M_ROWS * Dm * 2);
    unsigned short* WqkvT = (unsigned short*)alloc((size_t)N_QKV * Dm * 2);
    unsigned short* WoutT = (unsigned short*)alloc((size_t)Dm * Dm * 2);
    unsigned short* Qb    = (unsigned short*)alloc((size_t)M_ROWS * Dm * 2);
    unsigned short* Kb    = (unsigned short*)alloc((size_t)M_ROWS * Dm * 2);
    unsigned short* Vtb   = (unsigned short*)alloc((size_t)M_ROWS * Dm * 2);
    unsigned short* Ab    = (unsigned short*)alloc((size_t)M_ROWS * Dm * 2);

    const int nx = M_ROWS * Dm;                       // 8,388,608
    cvt_f32_bf16<<<(nx + 255) / 256, 256, 0, stream>>>(x, Xb, nx);
    transpose_bf16<<<(Dm * N_QKV + 255) / 256, 256, 0, stream>>>(Wqkv, WqkvT, Dm, N_QKV);
    transpose_bf16<<<(Dm * Dm + 255) / 256, 256, 0, stream>>>(Wout, WoutT, Dm, Dm);

    // QKV: 32 m-blocks (256 rows) x 48 n-groups (64 cols)
    qkv_gemm_kernel<<<32 * 48, 256, 0, stream>>>(Xb, WqkvT, bqkv, Qb, Kb, Vtb);
    // attention: 4*16*128 = 8192 waves, 8 waves/block
    attn_kernel<<<8192 / 8, 256, 0, stream>>>(Qb, Kb, Vtb, mask, head_out, Ab);
    // out-proj: 32 m-blocks x 16 n-groups
    out_gemm_kernel<<<32 * 16, 256, 0, stream>>>(Ab, WoutT, bout, out);
}
